// TGraphSAGE_12343736009440
// MI455X (gfx1250) — matrix operand order repaired and checked by reading, not compiled, not run on hardware
//
#include <hip/hip_runtime.h>

#define NN 100000   // nodes
#define NE 500000   // edges
#define DH 128      // hidden dim
#define TDIM 64     // time-encoding dim
// K for self GEMM = DH + TDIM = 192

typedef __attribute__((ext_vector_type(16))) __bf16 v16bf;
typedef __attribute__((ext_vector_type(8)))  float  v8f;
typedef __attribute__((ext_vector_type(4)))  float  v4f;

union BV {
  v16bf v;
  unsigned short h[16];
  uint4 q[2];
};

__device__ __forceinline__ unsigned short f2bf(float x) {
  return (unsigned short)(__float_as_uint(x) >> 16);   // truncate (hot path)
}
__device__ __forceinline__ unsigned short f2bf_rne(float x) {
  unsigned int u = __float_as_uint(x);
  u += 0x7fffu + ((u >> 16) & 1u);                     // round-nearest-even
  return (unsigned short)(u >> 16);
}

// ---------------------------------------------------------------------------
// Async copy of packed weights (global -> LDS) using the CDNA5 async-to-LDS
// path (ASYNCcnt-tracked), then wait on this wave's async counter.
// bytes must be a multiple of 16; blockDim.x == 128.
// ---------------------------------------------------------------------------
__device__ __forceinline__ void async_fill_lds(unsigned short* sdst,
                                               const unsigned short* gsrc,
                                               int bytes) {
  unsigned lds_base = (unsigned)(uintptr_t)(void*)sdst;   // low 32b = LDS offset
  unsigned long long gbase = (unsigned long long)(uintptr_t)gsrc;
  for (int off = (int)threadIdx.x * 16; off < bytes; off += 128 * 16) {
    unsigned la = lds_base + (unsigned)off;
    unsigned long long ga = gbase + (unsigned long long)off;
    asm volatile("global_load_async_to_lds_b128 %0, %1, off"
                 :: "v"(la), "v"(ga) : "memory");
  }
  asm volatile("s_wait_asynccnt 0" ::: "memory");
}

// ---------------------------------------------------------------------------
// Repack row-major f32 weight W[K][Ncols] into WMMA B-operand order (bf16):
// packed[kt][nt][lane][16]; lane -> col n=nt*16+(lane&15), kh=lane>>4:
// elem j<8 -> K = kt*32 + kh*8 + j, elem j>=8 -> K = kt*32 + 16 + kh*8 + (j-8).
// ---------------------------------------------------------------------------
__global__ void pack_weights(const float* __restrict__ W,
                             unsigned short* __restrict__ out,
                             int K, int Ncols) {
  int t = blockIdx.x * blockDim.x + threadIdx.x;
  int NT = Ncols >> 4;
  int total = (K >> 5) * NT * 32;
  if (t >= total) return;
  int lane = t & 31;
  int rest = t >> 5;
  int nt = rest % NT;
  int kt = rest / NT;
  int n = (nt << 4) + (lane & 15);
  int khalf = lane >> 4;
  unsigned short* o = out + (size_t)t * 16;
#pragma unroll
  for (int j = 0; j < 16; ++j) {
    int k = (kt << 5) + khalf * 8 + (j & 7) + ((j >> 3) << 4);
    o[j] = f2bf_rne(W[(size_t)k * Ncols + n]);
  }
}

// ---------------------------------------------------------------------------
// A-operand chunk loader: 16x32 bf16 fragment from a row-major f32 matrix.
// Lane holds row m = lane&15; khalf = lane>>4 selects K subset
// {kbase+kh*8+j} U {kbase+16+kh*8+j}, j=0..7 (per ISA 16-bit A layout).
// ---------------------------------------------------------------------------
__device__ __forceinline__ v16bf load_actA(const float* __restrict__ row,
                                           int kbase, int khalf, float scale) {
  BV u;
  const float* p0 = row + kbase + khalf * 8;
#pragma unroll
  for (int j = 0; j < 8; ++j) {
    u.h[j]     = f2bf(p0[j]      * scale);
    u.h[8 + j] = f2bf(p0[16 + j] * scale);
  }
  return u.v;
}

// B fragment from LDS-resident packed weights -> ds_load_b128 x2
__device__ __forceinline__ v16bf load_B_lds(const unsigned short* sw,
                                            int kt, int nt, int lane) {
  BV u;
  const uint4* p = (const uint4*)(sw + ((((unsigned)kt * 8 + nt) * 32 + lane) << 4));
  u.q[0] = p[0];
  u.q[1] = p[1];
  return u.v;
}

// ---------------------------------------------------------------------------
// combine_feats: per edge, out_src = [nfeat[src] | efeat], out_dst = [nfeat[dst] | efeat]
// one wave (32 lanes * float4 = 128 floats) per edge. NT stores (streaming).
// ---------------------------------------------------------------------------
__global__ void combine_feats(const float* __restrict__ nfeat,
                              const float* __restrict__ efeat,
                              const int* __restrict__ src,
                              const int* __restrict__ dst,
                              float* __restrict__ out_src,
                              float* __restrict__ out_dst) {
  int gid = blockIdx.x * blockDim.x + threadIdx.x;
  int e = gid >> 5;
  int lane = gid & 31;
  if (e >= NE) return;
  int s = src[e], d = dst[e];
  v4f vs, vd;
  if (lane < 16) {
    vs = ((const v4f*)(nfeat + (size_t)s * 64))[lane];
    vd = ((const v4f*)(nfeat + (size_t)d * 64))[lane];
  } else {
    vs = __builtin_nontemporal_load(((const v4f*)(efeat + (size_t)e * 64)) + (lane - 16));
    vd = vs;
  }
  __builtin_nontemporal_store(vs, ((v4f*)(out_src + (size_t)e * DH)) + lane);
  __builtin_nontemporal_store(vd, ((v4f*)(out_dst + (size_t)e * DH)) + lane);
}

// ---------------------------------------------------------------------------
// scatter_sum: segment sums via f32 atomics (resolve in L2; sum_n ~51MB fits).
// one wave per edge; 4 floats per lane per row. NT loads for streamed rows.
// ---------------------------------------------------------------------------
__global__ void scatter_sum(const float* __restrict__ prev_src,
                            const float* __restrict__ prev_dst,
                            const int* __restrict__ src,
                            const int* __restrict__ dst,
                            float* __restrict__ sum_n,
                            float* __restrict__ cnt) {
  int gid = blockIdx.x * blockDim.x + threadIdx.x;
  int e = gid >> 5;
  int lane = gid & 31;
  if (e >= NE) return;
  int s = src[e], d = dst[e];
  v4f fd = __builtin_nontemporal_load(((const v4f*)(prev_dst + (size_t)e * DH)) + lane);
  v4f fs = __builtin_nontemporal_load(((const v4f*)(prev_src + (size_t)e * DH)) + lane);
  float* ps = sum_n + (size_t)s * DH + lane * 4;
  float* pd = sum_n + (size_t)d * DH + lane * 4;
#pragma unroll
  for (int j = 0; j < 4; ++j) {
    __hip_atomic_fetch_add(ps + j, fd[j], __ATOMIC_RELAXED, __HIP_MEMORY_SCOPE_AGENT);
    __hip_atomic_fetch_add(pd + j, fs[j], __ATOMIC_RELAXED, __HIP_MEMORY_SCOPE_AGENT);
  }
  if (lane == 0) {
    __hip_atomic_fetch_add(cnt + s, 1.0f, __ATOMIC_RELAXED, __HIP_MEMORY_SCOPE_AGENT);
    __hip_atomic_fetch_add(cnt + d, 1.0f, __ATOMIC_RELAXED, __HIP_MEMORY_SCOPE_AGENT);
  }
}

// ---------------------------------------------------------------------------
// neigh_proj: proj = (sum_n / max(cnt,1)) @ W_neigh + b_neigh + b_self (folded)
// one wave per 16-node tile; W_neigh staged in LDS (async); 32 WMMA/wave.
// ---------------------------------------------------------------------------
__global__ void __launch_bounds__(128)
neigh_proj_kernel(const float* __restrict__ sum_n,
                  const float* __restrict__ cnt,
                  const unsigned short* __restrict__ pWn,
                  const float* __restrict__ b_neigh,
                  const float* __restrict__ b_self,
                  float* __restrict__ proj) {
  __shared__ unsigned short sW[4 * 8 * 32 * 16];     // 32 KB
  async_fill_lds(sW, pWn, 4 * 8 * 32 * 16 * 2);
  __syncthreads();

  int wave = (blockIdx.x * blockDim.x + threadIdx.x) >> 5;
  int lane = threadIdx.x & 31;
  int nbase = wave * 16;
  if (nbase >= NN) return;

  int mA = lane & 15;
  int khalf = lane >> 4;
  int node = nbase + mA;
  float rcp = 1.0f / fmaxf(cnt[node], 1.0f);
  const float* row = sum_n + (size_t)node * DH;

  v16bf a[4];
#pragma unroll
  for (int kt = 0; kt < 4; ++kt) a[kt] = load_actA(row, kt * 32, khalf, rcp);

  int nloc = lane & 15;
  int mAdd = (lane >> 4) * 8;   // C/D layout: lanes 16-31 hold rows 8..15
#pragma unroll
  for (int nt = 0; nt < 8; ++nt) {
    v8f acc = {};
#pragma unroll
    for (int kt = 0; kt < 4; ++kt) {
      v16bf b = load_B_lds(sW, kt, nt, lane);
      acc = __builtin_amdgcn_wmma_f32_16x16x32_bf16(false, a[kt], false, b,
                                                    (short)0, acc, false, false);
    }
    int n = nt * 16 + nloc;
    float bias = b_neigh[n] + b_self[n];
#pragma unroll
    for (int r = 0; r < 8; ++r) {
      proj[(size_t)(nbase + mAdd + r) * DH + n] = acc[r] + bias;
    }
  }
}

// ---------------------------------------------------------------------------
// edge_layer: per 16-edge tile, D=relu([prev|cos-te] @ W_self + proj[idx]).
// W_self staged in LDS (async). src/dst GEMMs share B + te fragments.
// 96 WMMA per wave; NT stores for the streaming output.
// ---------------------------------------------------------------------------
__global__ void __launch_bounds__(128)
edge_layer_kernel(const float* __restrict__ prev_src,
                  const float* __restrict__ prev_dst,
                  const int* __restrict__ src,
                  const int* __restrict__ dst,
                  const float* __restrict__ ts,
                  const float* __restrict__ omega,
                  const float* __restrict__ phase,
                  const unsigned short* __restrict__ pWs,
                  const float* __restrict__ proj,
                  float* __restrict__ out_src,
                  float* __restrict__ out_dst) {
  __shared__ unsigned short sW[6 * 8 * 32 * 16];     // 48 KB
  async_fill_lds(sW, pWs, 6 * 8 * 32 * 16 * 2);
  __syncthreads();

  int wave = (blockIdx.x * blockDim.x + threadIdx.x) >> 5;
  int lane = threadIdx.x & 31;
  int ebase = wave * 16;
  if (ebase >= NE) return;

  int mA = lane & 15;
  int khalf = lane >> 4;
  int eA = ebase + mA;
  const float* rs = prev_src + (size_t)eA * DH;
  const float* rd = prev_dst + (size_t)eA * DH;
  __builtin_prefetch(rs + 16 * DH, 0, 1);   // speculative stream-ahead
  __builtin_prefetch(rd + 16 * DH, 0, 1);

  float tval = ts[eA];
  v16bf aS[4], aD[4], aT[2];
#pragma unroll
  for (int kt = 0; kt < 4; ++kt) {
    aS[kt] = load_actA(rs, kt * 32, khalf, 1.0f);
    aD[kt] = load_actA(rd, kt * 32, khalf, 1.0f);
  }
  // cosine time-encoding fragments (K = 128..191 of the self GEMM)
#pragma unroll
  for (int c = 0; c < 2; ++c) {
    BV u;
    int kb = c * 32 + khalf * 8;
#pragma unroll
    for (int j = 0; j < 8; ++j) {
      int k0 = kb + j, k1 = kb + 16 + j;
      u.h[j]     = f2bf(__cosf(tval * omega[k0] + phase[k0]));
      u.h[8 + j] = f2bf(__cosf(tval * omega[k1] + phase[k1]));
    }
    aT[c] = u.v;
  }

  int nloc = lane & 15;
  int mAdd = (lane >> 4) * 8;
  int sIdx[8], dIdx[8];
#pragma unroll
  for (int r = 0; r < 8; ++r) {
    sIdx[r] = src[ebase + mAdd + r];
    dIdx[r] = dst[ebase + mAdd + r];
  }

#pragma unroll
  for (int nt = 0; nt < 8; ++nt) {
    v8f accS = {}, accD = {};
#pragma unroll
    for (int kt = 0; kt < 6; ++kt) {
      v16bf b = load_B_lds(sW, kt, nt, lane);
      v16bf As = (kt < 4) ? aS[kt] : aT[kt - 4];
      v16bf Ad = (kt < 4) ? aD[kt] : aT[kt - 4];
      accS = __builtin_amdgcn_wmma_f32_16x16x32_bf16(false, As, false, b,
                                                     (short)0, accS, false, false);
      accD = __builtin_amdgcn_wmma_f32_16x16x32_bf16(false, Ad, false, b,
                                                     (short)0, accD, false, false);
    }
    int n = nt * 16 + nloc;
#pragma unroll
    for (int r = 0; r < 8; ++r) {
      int e = ebase + mAdd + r;
      float vs = accS[r] + proj[(size_t)sIdx[r] * DH + n];
      float vd = accD[r] + proj[(size_t)dIdx[r] * DH + n];
      __builtin_nontemporal_store(fmaxf(vs, 0.0f), out_src + (size_t)e * DH + n);
      __builtin_nontemporal_store(fmaxf(vd, 0.0f), out_dst + (size_t)e * DH + n);
    }
  }
}

// ---------------------------------------------------------------------------
extern "C" void kernel_launch(void* const* d_in, const int* in_sizes, int n_in,
                              void* d_out, int out_size, void* d_ws, size_t ws_size,
                              hipStream_t stream) {
  (void)in_sizes; (void)n_in; (void)out_size; (void)ws_size;

  const float* nfeat   = (const float*)d_in[0];
  const float* efeat   = (const float*)d_in[1];
  const int*   src     = (const int*)d_in[2];
  const int*   dst     = (const int*)d_in[3];
  const float* ts      = (const float*)d_in[4];
  const float* W_self1  = (const float*)d_in[5];
  const float* b_self1  = (const float*)d_in[6];
  const float* W_neigh1 = (const float*)d_in[7];
  const float* b_neigh1 = (const float*)d_in[8];
  const float* omega1   = (const float*)d_in[9];
  const float* phase1   = (const float*)d_in[10];
  const float* W_self2  = (const float*)d_in[11];
  const float* b_self2  = (const float*)d_in[12];
  const float* W_neigh2 = (const float*)d_in[13];
  const float* b_neigh2 = (const float*)d_in[14];
  const float* omega2   = (const float*)d_in[15];
  const float* phase2   = (const float*)d_in[16];

  float* outS = (float*)d_out;                 // final src_feat [E,128]
  float* outD = outS + (size_t)NE * DH;        // final dst_feat [E,128]

  // workspace carve-up
  char* w = (char*)d_ws;
  float* bufS  = (float*)w; w += (size_t)NE * DH * 4;
  float* bufD  = (float*)w; w += (size_t)NE * DH * 4;
  float* sum_n = (float*)w; w += (size_t)NN * DH * 4;
  float* cnt   = (float*)w; w += (size_t)NN * 4;
  float* proj  = (float*)w; w += (size_t)NN * DH * 4;
  unsigned short* pWs1 = (unsigned short*)w; w += (size_t)6 * 8 * 32 * 16 * 2;
  unsigned short* pWn1 = (unsigned short*)w; w += (size_t)4 * 8 * 32 * 16 * 2;
  unsigned short* pWs2 = (unsigned short*)w; w += (size_t)6 * 8 * 32 * 16 * 2;
  unsigned short* pWn2 = (unsigned short*)w; w += (size_t)4 * 8 * 32 * 16 * 2;

  // 1) pack weights into WMMA B-operand layout (bf16)
  pack_weights<<<6, 256, 0, stream>>>(W_self1, pWs1, 192, 128);
  pack_weights<<<4, 256, 0, stream>>>(W_neigh1, pWn1, 128, 128);
  pack_weights<<<6, 256, 0, stream>>>(W_self2, pWs2, 192, 128);
  pack_weights<<<4, 256, 0, stream>>>(W_neigh2, pWn2, 128, 128);

  // 2) initial per-edge concat features -> ping buffer B (d_out)
  {
    int threads = NE * 32;
    combine_feats<<<(threads + 255) / 256, 256, 0, stream>>>(
        nfeat, efeat, src, dst, outS, outD);
  }

  const int scatterGrid = (NE * 32 + 255) / 256;
  const int projGrid    = ((NN / 16) + 3) / 4;      // 4 waves / 128-thread block
  const int edgeGrid    = ((NE / 16) + 3) / 4;

  // ---- layer 1: prev = d_out, out = ws bufs ----
  hipMemsetAsync(sum_n, 0, (size_t)NN * DH * 4, stream);
  hipMemsetAsync(cnt, 0, (size_t)NN * 4, stream);
  scatter_sum<<<scatterGrid, 256, 0, stream>>>(outS, outD, src, dst, sum_n, cnt);
  neigh_proj_kernel<<<projGrid, 128, 0, stream>>>(sum_n, cnt, pWn1, b_neigh1, b_self1, proj);
  edge_layer_kernel<<<edgeGrid, 128, 0, stream>>>(outS, outD, src, dst, ts,
                                                  omega1, phase1, pWs1, proj, bufS, bufD);

  // ---- layer 2: prev = ws bufs, out = d_out ----
  hipMemsetAsync(sum_n, 0, (size_t)NN * DH * 4, stream);
  hipMemsetAsync(cnt, 0, (size_t)NN * 4, stream);
  scatter_sum<<<scatterGrid, 256, 0, stream>>>(bufS, bufD, src, dst, sum_n, cnt);
  neigh_proj_kernel<<<projGrid, 128, 0, stream>>>(sum_n, cnt, pWn2, b_neigh2, b_self2, proj);
  edge_layer_kernel<<<edgeGrid, 128, 0, stream>>>(bufS, bufD, src, dst, ts,
                                                  omega2, phase2, pWs2, proj, outS, outD);
}
